// HGCN_4587025072674
// MI455X (gfx1250) — matrix operand order repaired and checked by reading, not compiled, not run on hardware
//
#include <hip/hip_runtime.h>
#include <hip/hip_bf16.h>
#include <math.h>

typedef __bf16 bf16_t;
typedef __attribute__((ext_vector_type(16))) __bf16 v16bf;
typedef __attribute__((ext_vector_type(8)))  __bf16 v8bf;
typedef __attribute__((ext_vector_type(8)))  float  v8f;

// Problem constants (reference: B=512, N=128, D=256, L=3, F=32)
constexpr int Bb = 512, Nn = 128, Dd = 256, Ll = 3, Ff = 32;
constexpr int ROWS = Bb * Nn;          // 65536 node rows
constexpr float EPSV = 1e-7f;
constexpr float MAX_NORM = 1.0f - 1e-5f;

// ---------------------------------------------------------------------------
// fp32 -> bf16 elementwise cast (for adjacency)
__global__ __launch_bounds__(256) void k_cast_bf16(const float* __restrict__ src,
                                                   bf16_t* __restrict__ dst, int n) {
    int i = blockIdx.x * 256 + threadIdx.x;
    if (i < n) dst[i] = (bf16_t)src[i];
}

// fp32 [rows x cols] -> bf16 transposed [cols x rows] (weights -> B^T layout)
__global__ __launch_bounds__(256) void k_transpose_bf16(const float* __restrict__ src,
                                                        bf16_t* __restrict__ dst,
                                                        int rows, int cols) {
    int i = blockIdx.x * 256 + threadIdx.x;
    if (i < rows * cols) {
        int r = i / cols, c = i % cols;
        dst[c * rows + r] = (bf16_t)src[i];
    }
}

// ---------------------------------------------------------------------------
// logmap0: t = artanh(||x||) * x / ||x||, one wave32 per 256-wide row
__device__ __forceinline__ float wave_reduce_add(float v) {
    #pragma unroll
    for (int off = 16; off > 0; off >>= 1) v += __shfl_xor(v, off, 32);
    return v;
}

__global__ __launch_bounds__(256) void k_logmap_f32(const float* __restrict__ x,
                                                    bf16_t* __restrict__ t) {
    int row  = blockIdx.x * 8 + (threadIdx.x >> 5);
    int lane = threadIdx.x & 31;
    const float* xr = x + (size_t)row * Dd + lane * 8;
    float v[8]; float ss = 0.f;
    #pragma unroll
    for (int e = 0; e < 8; e++) { v[e] = xr[e]; ss += v[e] * v[e]; }
    ss = wave_reduce_add(ss);
    float n  = sqrtf(ss);
    float nc = fminf(fmaxf(n, EPSV), 1.0f - 1e-7f);
    float at = 0.5f * (log1pf(nc) - log1pf(-nc));
    float sc = at / fmaxf(n, EPSV);
    union { v8bf v; bf16_t e[8]; } o;
    #pragma unroll
    for (int e = 0; e < 8; e++) o.e[e] = (bf16_t)(v[e] * sc);
    *(v8bf*)(t + (size_t)row * Dd + lane * 8) = o.v;
}

__global__ __launch_bounds__(256) void k_logmap_bf16(const bf16_t* __restrict__ x,
                                                     bf16_t* __restrict__ t) {
    int row  = blockIdx.x * 8 + (threadIdx.x >> 5);
    int lane = threadIdx.x & 31;
    union { v8bf v; bf16_t e[8]; } in;
    in.v = *(const v8bf*)(x + (size_t)row * Dd + lane * 8);
    float v[8]; float ss = 0.f;
    #pragma unroll
    for (int e = 0; e < 8; e++) { v[e] = (float)in.e[e]; ss += v[e] * v[e]; }
    ss = wave_reduce_add(ss);
    float n  = sqrtf(ss);
    float nc = fminf(fmaxf(n, EPSV), 1.0f - 1e-7f);
    float at = 0.5f * (log1pf(nc) - log1pf(-nc));
    float sc = at / fmaxf(n, EPSV);
    union { v8bf v; bf16_t e[8]; } o;
    #pragma unroll
    for (int e = 0; e < 8; e++) o.e[e] = (bf16_t)(v[e] * sc);
    *(v8bf*)(t + (size_t)row * Dd + lane * 8) = o.v;
}

// combine: x = expmap0(relu(h + s)) with Poincare-ball projection
__global__ __launch_bounds__(256) void k_combine(const bf16_t* __restrict__ h,
                                                 const bf16_t* __restrict__ s,
                                                 bf16_t* __restrict__ x) {
    int row  = blockIdx.x * 8 + (threadIdx.x >> 5);
    int lane = threadIdx.x & 31;
    union { v8bf v; bf16_t e[8]; } hv, sv;
    hv.v = *(const v8bf*)(h + (size_t)row * Dd + lane * 8);
    sv.v = *(const v8bf*)(s + (size_t)row * Dd + lane * 8);
    float u[8]; float ss = 0.f;
    #pragma unroll
    for (int e = 0; e < 8; e++) {
        float y = (float)hv.e[e] + (float)sv.e[e];
        y = y > 0.f ? y : 0.f;
        u[e] = y; ss += y * y;
    }
    ss = wave_reduce_add(ss);
    float n  = fmaxf(sqrtf(ss), EPSV);
    float th = tanhf(n);
    float f  = th / n;                       // exp_0 scale
    if (th > MAX_NORM) f = MAX_NORM / n;     // ball projection
    union { v8bf v; bf16_t e[8]; } o;
    #pragma unroll
    for (int e = 0; e < 8; e++) o.e[e] = (bf16_t)(u[e] * f);
    *(v8bf*)(x + (size_t)row * Dd + lane * 8) = o.v;
}

// ---------------------------------------------------------------------------
// Generic bf16 WMMA GEMM: C[M,N] = A[M,K] * Bt[N,K]^T (+bias) with epilogues.
// Block tile 128x64, BK=64 double-buffered in LDS; 8 waves, each a 32x32
// sub-tile -> 8 v_wmma_f32_16x16x32_bf16 per barrier.
// mode 0: store bf16 row-major (no act)
// mode 1: relu, store bf16 transposed-per-batch [b][n][node] (msg^T producer)
// mode 2: fp32 row-major * node_mask (output head)
__global__ __launch_bounds__(256) void k_gemm(const bf16_t* __restrict__ A, int lda,
                                              const bf16_t* __restrict__ Bt,
                                              int bt_batch_stride,
                                              const float* __restrict__ bias,
                                              const float* __restrict__ mask,
                                              void* __restrict__ out,
                                              int M, int K, int Ncols, int mode) {
    constexpr int BM = 128, BN = 64, BK = 64, LDP = 72; // 144B row stride: no bank conflicts
    __shared__ bf16_t As[2][BM][LDP];   // 36864 B
    __shared__ bf16_t Bs[2][BN][LDP];   // 18432 B

    const int tid  = threadIdx.x;
    const int m0   = blockIdx.x * BM;
    const int n0   = blockIdx.y * BN;
    const bf16_t* Btb = Bt + (size_t)(m0 >> 7) * (size_t)bt_batch_stride;

    const int w    = tid >> 5;
    const int lane = tid & 31;
    const int wm   = (w & 3) * 32;   // wave M offset in block tile
    const int wn   = (w >> 2) * 32;  // wave N offset
    const int lm   = lane & 15;
    const int kh   = lane >> 4;

    // staging decomposition (8 chunks of 8 bf16 per 64-wide row)
    const int arow0 = tid >> 3;      // 0..31  (+32*it covers 128 rows, 4 iters)
    const int ac    = (tid & 7) * 8; // chunk offset in k
    const int brow0 = tid >> 3;      // 0..31  (+32*it covers 64 rows, 2 iters)

    auto stage = [&](int kb, int buf) {
        const int kbase = kb * BK;
        #pragma unroll
        for (int it = 0; it < 4; ++it) {
            int row = arow0 + it * 32;
            *(v8bf*)&As[buf][row][ac] =
                *(const v8bf*)(A + (size_t)(m0 + row) * lda + kbase + ac);
        }
        #pragma unroll
        for (int it = 0; it < 2; ++it) {
            int row = brow0 + it * 32;
            v8bf val;
            if (n0 + row < Ncols) {
                val = *(const v8bf*)(Btb + (size_t)(n0 + row) * K + kbase + ac);
            } else {
                union { v8bf v; unsigned u[4]; } z;
                z.u[0] = z.u[1] = z.u[2] = z.u[3] = 0u;
                val = z.v;
            }
            *(v8bf*)&Bs[buf][row][ac] = val;
        }
    };

    v8f acc[2][2];
    #pragma unroll
    for (int i = 0; i < 2; i++)
        #pragma unroll
        for (int j = 0; j < 2; j++)
            #pragma unroll
            for (int e = 0; e < 8; e++) acc[i][j][e] = 0.f;

    const int nkb = K / BK;
    stage(0, 0);
    __syncthreads();
    for (int kb = 0; kb < nkb; ++kb) {
        const int cur = kb & 1;
        if (kb + 1 < nkb) stage(kb + 1, cur ^ 1);  // overlap next slice with WMMA

        #pragma unroll
        for (int ks = 0; ks < 2; ++ks) {           // two 32-wide K sub-steps
            const int ko = ks * 32;
            v16bf afrag[2], bfrag[2];
            #pragma unroll
            for (int i = 0; i < 2; i++) {
                int r = wm + i * 16 + lm;
                v8bf lo = *(const v8bf*)&As[cur][r][ko + kh * 8];
                v8bf hi = *(const v8bf*)&As[cur][r][ko + 16 + kh * 8];
                afrag[i] = __builtin_shufflevector(lo, hi, 0, 1, 2, 3, 4, 5, 6, 7,
                                                   8, 9, 10, 11, 12, 13, 14, 15);
            }
            #pragma unroll
            for (int j = 0; j < 2; j++) {
                int r = wn + j * 16 + lm;
                v8bf lo = *(const v8bf*)&Bs[cur][r][ko + kh * 8];
                v8bf hi = *(const v8bf*)&Bs[cur][r][ko + 16 + kh * 8];
                bfrag[j] = __builtin_shufflevector(lo, hi, 0, 1, 2, 3, 4, 5, 6, 7,
                                                   8, 9, 10, 11, 12, 13, 14, 15);
            }
            #pragma unroll
            for (int i = 0; i < 2; i++)
                #pragma unroll
                for (int j = 0; j < 2; j++)
                    acc[i][j] = __builtin_amdgcn_wmma_f32_16x16x32_bf16(
                        false, afrag[i], false, bfrag[j], (short)0, acc[i][j],
                        false, false);
        }
        __syncthreads();  // reads(cur) done before kb+1 overwrites it; writes(nxt) visible
    }

    // Epilogue: C layout — lane column n = lm, rows m = r + 8*kh
    #pragma unroll
    for (int i = 0; i < 2; i++) {
        int base_m = m0 + wm + i * 16 + kh * 8;
        #pragma unroll
        for (int j = 0; j < 2; j++) {
            int gn = n0 + wn + j * 16 + lm;
            if (gn >= Ncols) continue;
            float bv = bias ? bias[gn] : 0.f;
            if (mode == 0) {
                bf16_t* o = (bf16_t*)out;
                #pragma unroll
                for (int r = 0; r < 8; r++) {
                    float v = acc[i][j][r] + bv;
                    o[(size_t)(base_m + r) * Ncols + gn] = (bf16_t)v;
                }
            } else if (mode == 1) {
                // relu + store transposed per batch of 128 nodes: [b][gn][node]
                bf16_t* o  = (bf16_t*)out;
                int batch  = base_m >> 7;
                int node   = base_m & 127;
                union { v8bf v; bf16_t e[8]; } pk;
                #pragma unroll
                for (int r = 0; r < 8; r++) {
                    float v = acc[i][j][r] + bv;
                    pk.e[r] = (bf16_t)(v > 0.f ? v : 0.f);
                }
                *(v8bf*)(o + (size_t)batch * ((size_t)Ncols * 128) +
                         (size_t)gn * 128 + node) = pk.v;
            } else {
                float* o = (float*)out;
                #pragma unroll
                for (int r = 0; r < 8; r++) {
                    int m   = base_m + r;
                    float v = (acc[i][j][r] + bv) * mask[m];
                    o[(size_t)m * Ncols + gn] = v;
                }
            }
        }
    }
}

// ---------------------------------------------------------------------------
extern "C" void kernel_launch(void* const* d_in, const int* in_sizes, int n_in,
                              void* d_out, int out_size, void* d_ws, size_t ws_size,
                              hipStream_t stream) {
    (void)in_sizes; (void)n_in; (void)out_size; (void)ws_size;
    const float* x        = (const float*)d_in[0];
    const float* adj      = (const float*)d_in[1];
    const float* nodemask = (const float*)d_in[2];
    const float* W        = (const float*)d_in[3];
    const float* b        = (const float*)d_in[4];
    const float* Wmsg     = (const float*)d_in[5];
    const float* bmsg     = (const float*)d_in[6];
    const float* Wsum     = (const float*)d_in[7];
    const float* bsum     = (const float*)d_in[8];
    const float* Wout     = (const float*)d_in[9];
    const float* bout     = (const float*)d_in[10];

    char* ws = (char*)d_ws;
    size_t off = 0;
    auto alloc = [&](size_t bytes) {
        void* p = ws + off;
        off += (bytes + 255) & ~(size_t)255;
        return p;
    };
    bf16_t* adjb  = (bf16_t*)alloc((size_t)Bb * Nn * Nn * 2);  // 16 MB
    bf16_t* Wt    = (bf16_t*)alloc((size_t)Ll * Dd * Dd * 2);
    bf16_t* WmsgT = (bf16_t*)alloc((size_t)Ll * Dd * Dd * 2);
    bf16_t* WsumT = (bf16_t*)alloc((size_t)Ll * Dd * Dd * 2);
    bf16_t* WoutT = (bf16_t*)alloc((size_t)Ff * Dd * 2);
    bf16_t* xbuf  = (bf16_t*)alloc((size_t)ROWS * Dd * 2);     // 32 MB each
    bf16_t* tbuf  = (bf16_t*)alloc((size_t)ROWS * Dd * 2);
    bf16_t* hbuf  = (bf16_t*)alloc((size_t)ROWS * Dd * 2);
    bf16_t* mbuf  = (bf16_t*)alloc((size_t)ROWS * Dd * 2);     // msg^T

    // One-time per launch: convert adj + transpose/convert weights to bf16
    {
        int nadj = Bb * Nn * Nn;
        k_cast_bf16<<<(nadj + 255) / 256, 256, 0, stream>>>(adj, adjb, nadj);
        int nW = Dd * Dd;
        for (int l = 0; l < Ll; l++) {
            k_transpose_bf16<<<(nW + 255) / 256, 256, 0, stream>>>(
                W + (size_t)l * nW, Wt + (size_t)l * nW, Dd, Dd);
            k_transpose_bf16<<<(nW + 255) / 256, 256, 0, stream>>>(
                Wmsg + (size_t)l * nW, WmsgT + (size_t)l * nW, Dd, Dd);
            k_transpose_bf16<<<(nW + 255) / 256, 256, 0, stream>>>(
                Wsum + (size_t)l * nW, WsumT + (size_t)l * nW, Dd, Dd);
        }
        k_transpose_bf16<<<(Dd * Ff + 255) / 256, 256, 0, stream>>>(Wout, WoutT, Dd, Ff);
    }

    dim3 blk(256);
    dim3 g_dense(ROWS / 128, Dd / 64);  // 512 x 4
    dim3 g_head(ROWS / 128, 1);
    dim3 g_row(ROWS / 8);

    for (int l = 0; l < Ll; l++) {
        // h_t = logmap0(x)
        if (l == 0) k_logmap_f32<<<g_row, blk, 0, stream>>>(x, tbuf);
        else        k_logmap_bf16<<<g_row, blk, 0, stream>>>(xbuf, tbuf);
        // h = h_t @ W[l] + b[l]
        k_gemm<<<g_dense, blk, 0, stream>>>(tbuf, Dd, Wt + (size_t)l * Dd * Dd, 0,
                                            b + (size_t)l * Dd, nullptr, hbuf,
                                            ROWS, Dd, Dd, 0);
        // msg^T = relu(h @ Wmsg[l] + bmsg[l]) stored [b][d][node]
        k_gemm<<<g_dense, blk, 0, stream>>>(hbuf, Dd, WmsgT + (size_t)l * Dd * Dd, 0,
                                            bmsg + (size_t)l * Dd, nullptr, mbuf,
                                            ROWS, Dd, Dd, 1);
        // agg = adj @ msg  (per batch; adj flat [b*N+m][N], msg^T as batched B^T)
        k_gemm<<<g_dense, blk, 0, stream>>>(adjb, Nn, mbuf, Dd * Nn,
                                            nullptr, nullptr, xbuf,
                                            ROWS, Nn, Dd, 0);
        // s = agg @ Wsum[l] + bsum[l]
        k_gemm<<<g_dense, blk, 0, stream>>>(xbuf, Dd, WsumT + (size_t)l * Dd * Dd, 0,
                                            bsum + (size_t)l * Dd, nullptr, tbuf,
                                            ROWS, Dd, Dd, 0);
        // x = expmap0(relu(h + s))
        k_combine<<<g_row, blk, 0, stream>>>(hbuf, tbuf, xbuf);
    }
    // out = (logmap0(x) @ Wout + bout) * node_mask
    k_logmap_bf16<<<g_row, blk, 0, stream>>>(xbuf, tbuf);
    k_gemm<<<g_head, blk, 0, stream>>>(tbuf, Dd, WoutT, 0, bout, nodemask,
                                       d_out, ROWS, Dd, Ff, 2);
}
